// MoE_layer_fastermoe_46712064311452
// MI455X (gfx1250) — compile-verified, matrix operand order checked by reading, not run on hardware
//
#include <hip/hip_runtime.h>
#include <hip/hip_bf16.h>
#include <math.h>

// Problem constants (match the reference)
#define T_TOK 16384
#define H_DIM 2048
#define F_DIM 4096
#define E_NUM 8
#define CAP   (T_TOK / E_NUM)   // 2048 tokens per expert

// Tiling: block tile 128(M) x 256(N) x 64(K), 8 wave32s each owning 64x64.
#define BM 128
#define BN 256
#define BK 64
#define LSTRIDE 72                    // BK halves + 8 halves (16B) pad, keeps 16B alignment
#define ATILE_HALVES (BM * LSTRIDE)   // 9216 halves
#define BTILE_HALVES (BN * LSTRIDE)   // 18432 halves

typedef _Float16 v16h __attribute__((ext_vector_type(16)));
typedef _Float16 v8h  __attribute__((ext_vector_type(8)));
typedef float    v8f  __attribute__((ext_vector_type(8)));

// Build a 16x32 f16 WMMA operand fragment for this lane from an LDS tile.
// Per CDNA5 ISA: lane L holds row (L%16); halves 0..7 are K = base..base+7,
// halves 8..15 are K = base+16..base+23, base = 8*(L/16).
__device__ __forceinline__ v16h load_frag(const _Float16* p) {
    v8h lo = *(const v8h*)(p);
    v8h hi = *(const v8h*)(p + 16);
    return __builtin_shufflevector(lo, hi,
        0, 1, 2, 3, 4, 5, 6, 7, 8, 9, 10, 11, 12, 13, 14, 15);
}

__device__ __forceinline__ v8h cvt8(float4 f0, float4 f1) {
    v8h h;
    h[0] = (_Float16)f0.x; h[1] = (_Float16)f0.y;
    h[2] = (_Float16)f0.z; h[3] = (_Float16)f0.w;
    h[4] = (_Float16)f1.x; h[5] = (_Float16)f1.y;
    h[6] = (_Float16)f1.z; h[7] = (_Float16)f1.w;
    return h;
}

__device__ __forceinline__ float gelu_tanh(float x) {
    float x3 = x * x * x;
    float t  = tanhf(0.7978845608028654f * (x + 0.044715f * x3));
    return 0.5f * x * (1.0f + t);
}

// C(M x N) = A(M x K) * B(N x K)^T  (B rows are output columns; K contiguous)
// A_HALF:        A source is f16 (else fp32, converted on load)
// GELU_HALF_OUT: apply tanh-GELU and store f16 (else store raw fp32)
template <bool A_HALF, bool GELU_HALF_OUT>
__global__ __launch_bounds__(256) void moe_gemm_kernel(
    const void* __restrict__ Av, const float* __restrict__ Bg,
    void* __restrict__ Outv,
    int M, int N, int K,
    size_t aStride, size_t bStride, size_t oStride)
{
    // [buf][A|B] double-buffered: 2*(9216+18432) halves = 110,592 bytes
    __shared__ _Float16 smem[2 * (ATILE_HALVES + BTILE_HALVES)];

    const int e   = blockIdx.z;
    const int bm  = blockIdx.y * BM;
    const int bn  = blockIdx.x * BN;
    const int tid = threadIdx.x;

    const int lane   = tid & 31;
    const int wave   = tid >> 5;
    const int wm     = (wave & 1) * 64;   // wave row offset within block tile
    const int wn     = (wave >> 1) * 64;  // wave col offset within block tile
    const int half16 = lane >> 4;         // selects K sub-lane group
    const int lrow   = lane & 15;         // row/col within 16x16 tile

    // Per-expert / per-block base pointers
    const float*    Af = nullptr;
    const _Float16* Ah = nullptr;
    if constexpr (A_HALF) Ah = (const _Float16*)Av + e * aStride + (size_t)bm * K;
    else                  Af = (const float*)Av    + e * aStride + (size_t)bm * K;
    const float* Bf = Bg + e * bStride + (size_t)bn * K;

    // Cooperative loader mapping:
    //   A tile 128x64: thread (tid&127) row, 32 contiguous K elems at (tid>>7)*32
    //   B tile 256x64: thread tid -> one row, all 64 contiguous K elems
    const int lr = tid & 127;
    const int lc = (tid >> 7) * 32;

    v8h ra[4];   // 32 halves of A
    v8h rb[8];   // 64 halves of B

    auto loadA = [&](int kt) {
        if constexpr (A_HALF) {
            const _Float16* p = Ah + (size_t)lr * K + kt * BK + lc;
            #pragma unroll
            for (int q = 0; q < 4; ++q) ra[q] = *(const v8h*)(p + q * 8);
        } else {
            const float* p = Af + (size_t)lr * K + kt * BK + lc;
            #pragma unroll
            for (int q = 0; q < 4; ++q) {
                float4 f0 = *(const float4*)(p + q * 8);
                float4 f1 = *(const float4*)(p + q * 8 + 4);
                ra[q] = cvt8(f0, f1);
            }
        }
    };
    auto loadB = [&](int kt) {
        const float* p = Bf + (size_t)tid * K + kt * BK;
        #pragma unroll
        for (int q = 0; q < 8; ++q) {
            float4 f0 = *(const float4*)(p + q * 8);
            float4 f1 = *(const float4*)(p + q * 8 + 4);
            rb[q] = cvt8(f0, f1);
        }
    };
    auto stash = [&](int buf) {
        _Float16* sA = smem + buf * (ATILE_HALVES + BTILE_HALVES);
        _Float16* sB = sA + ATILE_HALVES;
        _Float16* pa = sA + lr * LSTRIDE + lc;
        _Float16* pb = sB + tid * LSTRIDE;
        #pragma unroll
        for (int q = 0; q < 4; ++q) *(v8h*)(pa + q * 8) = ra[q];
        #pragma unroll
        for (int q = 0; q < 8; ++q) *(v8h*)(pb + q * 8) = rb[q];
    };

    v8f acc[4][4] = {};

    auto compute = [&](int buf) {
        const _Float16* sA = smem + buf * (ATILE_HALVES + BTILE_HALVES);
        const _Float16* sB = sA + ATILE_HALVES;
        #pragma unroll
        for (int kk = 0; kk < BK; kk += 32) {
            v16h af[4], bf4[4];
            #pragma unroll
            for (int i = 0; i < 4; ++i)
                af[i] = load_frag(sA + (wm + i * 16 + lrow) * LSTRIDE + kk + 8 * half16);
            #pragma unroll
            for (int j = 0; j < 4; ++j)
                bf4[j] = load_frag(sB + (wn + j * 16 + lrow) * LSTRIDE + kk + 8 * half16);
            #pragma unroll
            for (int i = 0; i < 4; ++i)
                #pragma unroll
                for (int j = 0; j < 4; ++j)
                    acc[i][j] = __builtin_amdgcn_wmma_f32_16x16x32_f16(
                        false, af[i], false, bf4[j],
                        (short)0, acc[i][j], false, false);
        }
    };

    const int numT = K / BK;
    loadA(0); loadB(0);
    stash(0);
    __syncthreads();

    for (int t = 0; t < numT; ++t) {
        const int cur = t & 1;
        if (t + 1 < numT) { loadA(t + 1); loadB(t + 1); }
        compute(cur);
        if (t + 1 < numT) {
            stash(1 - cur);       // other buffer: safe while others still compute on cur
            __syncthreads();      // buf (1-cur) ready; buf cur free to overwrite next iter
        }
    }

    // Epilogue. C/D layout: lane L, vgpr r -> row = r + 8*(L/16), col = L%16.
    if constexpr (GELU_HALF_OUT) {
        _Float16* Oh = (_Float16*)Outv + e * oStride + (size_t)bm * N;
        #pragma unroll
        for (int i = 0; i < 4; ++i)
            #pragma unroll
            for (int j = 0; j < 4; ++j)
                #pragma unroll
                for (int r = 0; r < 8; ++r) {
                    const int row = wm + i * 16 + half16 * 8 + r;
                    const int col = bn + wn + j * 16 + lrow;
                    Oh[(size_t)row * N + col] = (_Float16)gelu_tanh(acc[i][j][r]);
                }
    } else {
        float* Of = (float*)Outv + e * oStride + (size_t)bm * N;
        #pragma unroll
        for (int i = 0; i < 4; ++i)
            #pragma unroll
            for (int j = 0; j < 4; ++j)
                #pragma unroll
                for (int r = 0; r < 8; ++r) {
                    const int row = wm + i * 16 + half16 * 8 + r;
                    const int col = bn + wn + j * 16 + lrow;
                    Of[(size_t)row * N + col] = acc[i][j][r];
                }
    }
}

extern "C" void kernel_launch(void* const* d_in, const int* in_sizes, int n_in,
                              void* d_out, int out_size, void* d_ws, size_t ws_size,
                              hipStream_t stream) {
    (void)in_sizes; (void)n_in; (void)out_size; (void)ws_size;

    const float* x  = (const float*)d_in[0];   // (T, H) fp32, pre-grouped by expert
    const float* w1 = (const float*)d_in[1];   // (E, F, H) fp32
    const float* w2 = (const float*)d_in[2];   // (E, H, F) fp32
    // d_in[3] = splits: static balanced CAP per expert, unused at runtime.

    float*    out = (float*)d_out;             // (T, H) fp32
    _Float16* act = (_Float16*)d_ws;           // (E, CAP, F) f16 intermediate (~134 MB)

    dim3 blk(256);

    // GEMM1 + GELU: h = gelu(x @ W1^T) -> f16 activations in workspace
    dim3 g1(F_DIM / BN, CAP / BM, E_NUM);
    moe_gemm_kernel<false, true><<<g1, blk, 0, stream>>>(
        (const void*)x, w1, (void*)act,
        CAP, F_DIM, H_DIM,
        (size_t)CAP * H_DIM, (size_t)F_DIM * H_DIM, (size_t)CAP * F_DIM);

    // GEMM2: out = a @ W2^T -> fp32
    dim3 g2(H_DIM / BN, CAP / BM, E_NUM);
    moe_gemm_kernel<true, false><<<g2, blk, 0, stream>>>(
        (const void*)act, w2, (void*)out,
        CAP, H_DIM, F_DIM,
        (size_t)CAP * F_DIM, (size_t)H_DIM * F_DIM, (size_t)CAP * H_DIM);
}
